// Classifier_71356586655750
// MI455X (gfx1250) — matrix operand order repaired and checked by reading, not compiled, not run on hardware
//
#include <hip/hip_runtime.h>

// ---------------------------------------------------------------------------
// Meta-learning inner loop on CDNA5 (gfx1250), one episode per workgroup.
// - All large matmuls: v_wmma_f32_16x16x32_f16 (f32 accumulate), f32 master
//   weights in LDS, f16 shadows as WMMA operands.
// - f32 weight staging uses global_load_async_to_lds_b128 (ASYNCcnt).
// - All f16 LDS rows padded +16B -> conflict-free ds_load_b128 fragments.
// ---------------------------------------------------------------------------

typedef __attribute__((ext_vector_type(16))) _Float16 v16h;
typedef __attribute__((ext_vector_type(8)))  float    v8f;

constexpr int E_Q    = 75;     // queries per episode
constexpr int S_SUP  = 25;     // support per episode
constexpr int SP     = 32;     // padded support rows
constexpr int QP     = 80;     // padded query rows
constexpr int FEAT   = 64;
constexpr int HID    = 128;
constexpr int WAY    = 5;
constexpr int WP     = 16;     // padded WAY
constexpr int ITERS  = 10;
constexpr float LR   = 0.01f;

// Padded f16 row strides (each +8 halves = +16B: keeps b128 alignment and
// makes per-lane row starts cover distinct LDS bank groups).
constexpr int LF = FEAT + 8;   // 72  (144 B rows)
constexpr int LH = HID  + 8;   // 136 (272 B rows)
constexpr int LS = SP   + 8;   // 40  (80 B rows)

struct SharedMem {
  float     W1f[HID][FEAT];    // master weights (f32, contiguous: async target)
  _Float16  W1h[HID][LF];      // f16 shadow (WMMA B operand)
  float     W2f[WAY][HID];     // master (contiguous: async target)
  _Float16  W2h[WP][LH];       // rows >= WAY zero
  float     b1f[HID];          // async target
  float     b2f[WP];
  _Float16  Xs [SP][LF];       // support feats (rows >= S zero)
  _Float16  XsT[FEAT][LS];     // transposed (B operand of dW1)
  _Float16  Xq [QP][LF];       // query feats (rows >= Q zero)
  int       tgt[SP];
  union {
    struct {
      _Float16 hS  [SP][LH];      // relu activations (support)
      _Float16 dz1T[HID][LS];     // grad wrt z1, transposed (A of dW1)
      float    logits[SP][WP];
      float    dlog  [SP][WP];    // dLoss/dlogits (incl 1/S)
    } tr;
    _Float16 hQ[QP][LH];          // query activations (reuses dead space)
  } u;
};

struct __align__(16) U128 { unsigned int x[4]; };
union Frag16 { v16h h; U128 q[2]; };

// ---- CDNA5 async global->LDS staging (ASYNCcnt) ----
__device__ __forceinline__ unsigned lds_off(const void* p) {
  // generic LDS pointer: low 32 bits are the LDS byte offset (ISA 10.2)
  return (unsigned)(unsigned long long)p;
}
__device__ __forceinline__ void async_b128(unsigned lds_addr, const void* gsrc) {
#if defined(__gfx1250__)
  asm volatile("global_load_async_to_lds_b128 %0, %1, off"
               :: "v"(lds_addr), "v"(gsrc) : "memory");
#endif
}
__device__ __forceinline__ void wait_async0() {
#if defined(__gfx1250__)
  asm volatile("s_wait_asynccnt 0x0" ::: "memory");
#endif
}

// A fragment (16x32 f16, row-major, K-contiguous).  ISA 7.12.2:
// lane = g*16+m ; VGPR v holds halves K = (v<4?0:16) + 8g + 2(v&3) {+0,+1}
// -> two contiguous 16B groups at half-offsets g*8 and 16+g*8.
__device__ __forceinline__ v16h load_a(const _Float16* A, int lda,
                                       int m0, int k0, int lane) {
  const int g = lane >> 4, m = lane & 15;
  const _Float16* p = A + (m0 + m) * lda + k0 + g * 8;
  Frag16 f;
  f.q[0] = *(const U128*)(p);
  f.q[1] = *(const U128*)(p + 16);
  return f.h;
}

// B fragment (32x16 f16) from Bt (N-major, K-contiguous), Bt[n][k] = B[k][n].
// lane = g*16+n ; halves 0..7 hold K = 8g + h, halves 8..15 hold K = 16 + 8g + (h-8).
__device__ __forceinline__ v16h load_b(const _Float16* Bt, int ldb,
                                       int n0, int k0, int lane) {
  const int g = lane >> 4, n = lane & 15;
  const _Float16* p = Bt + (n0 + n) * ldb + k0 + g * 8;
  Frag16 f;
  f.q[0] = *(const U128*)(p);
  f.q[1] = *(const U128*)(p + 16);
  return f.h;
}

__device__ __forceinline__ v8f wmma16(v16h a, v16h b, v8f c) {
  return __builtin_amdgcn_wmma_f32_16x16x32_f16(
      false, a, false, b, (short)0, c, false, false);
}

__global__ __launch_bounds__(128)
void episode_train_eval_kernel(const float* __restrict__ qf,
                               const float* __restrict__ sf,
                               const int*   __restrict__ st,
                               const float* __restrict__ W1g,
                               const float* __restrict__ b1g,
                               const float* __restrict__ W2g,
                               const float* __restrict__ b2g,
                               float*       __restrict__ out) {
  extern __shared__ char smem_raw[];
  SharedMem& sm = *reinterpret_cast<SharedMem*>(smem_raw);

  const int e    = blockIdx.x;
  const int tid  = threadIdx.x;           // 128 threads = 4 waves
  const int lane = tid & 31;
  const int wave = tid >> 5;

  // ------------- Stage 1: fire async copies of the f32 weights -------------
  {
    const unsigned lw1 = lds_off(&sm.W1f[0][0]);
    const char* g1 = (const char*)(W1g + (size_t)e * HID * FEAT);
    for (int i = tid; i < HID * FEAT * 4 / 16; i += 128)   // 2048 chunks
      async_b128(lw1 + i * 16, g1 + i * 16);
    const unsigned lw2 = lds_off(&sm.W2f[0][0]);
    const char* g2 = (const char*)(W2g + (size_t)e * WAY * HID);
    for (int i = tid; i < WAY * HID * 4 / 16; i += 128)    // 160 chunks
      async_b128(lw2 + i * 16, g2 + i * 16);
    const unsigned lb1 = lds_off(&sm.b1f[0]);
    const char* g3 = (const char*)(b1g + (size_t)e * HID);
    if (tid < HID * 4 / 16)                                // 32 chunks
      async_b128(lb1 + tid * 16, g3 + tid * 16);
  }

  // ------------- Stage 2: overlap convert-staging of features -------------
  for (int i = tid; i < SP * FEAT; i += 128) {
    const int s = i / FEAT, f = i % FEAT;
    const float v = (s < S_SUP) ? sf[(size_t)e * S_SUP * FEAT + s * FEAT + f] : 0.f;
    const _Float16 hv = (_Float16)v;
    sm.Xs[s][f]  = hv;
    sm.XsT[f][s] = hv;
  }
  for (int i = tid; i < QP * FEAT; i += 128) {
    const int q = i / FEAT, f = i % FEAT;
    const float v = (q < E_Q) ? qf[(size_t)e * E_Q * FEAT + q * FEAT + f] : 0.f;
    sm.Xq[q][f] = (_Float16)v;
  }
  if (tid < SP) sm.tgt[tid] = (tid < S_SUP) ? st[(size_t)e * S_SUP + tid] : 0;
  if (tid < WP) sm.b2f[tid] = (tid < WAY) ? b2g[(size_t)e * WAY + tid] : 0.f;
  // zero the padded W2h rows (w >= WAY)
  for (int i = tid; i < (WP - WAY) * HID; i += 128)
    sm.W2h[WAY + i / HID][i % HID] = (_Float16)0.f;

  // ------------- Stage 3: wait for async, build f16 shadows -------------
  wait_async0();
  __syncthreads();
  for (int i = tid; i < HID * FEAT / 2; i += 128) {        // 4096 float2
    const float2 v = *((const float2*)&sm.W1f[0][0] + i);
    const int r = (2 * i) >> 6, c = (2 * i) & 63;
    sm.W1h[r][c]     = (_Float16)v.x;
    sm.W1h[r][c + 1] = (_Float16)v.y;
  }
  for (int i = tid; i < WAY * HID / 2; i += 128) {         // 320 float2
    const float2 v = *((const float2*)&sm.W2f[0][0] + i);
    const int r = (2 * i) >> 7, c = (2 * i) & 127;
    sm.W2h[r][c]     = (_Float16)v.x;
    sm.W2h[r][c + 1] = (_Float16)v.y;
  }
  __syncthreads();

  // ---------------- 10 SGD steps, fully LDS-resident ----------------
  for (int it = 0; it < ITERS; ++it) {
    // (1) hS = relu(Xs @ W1^T + b1)  : 2x8 tiles, K=64  (WMMA)
    for (int t = wave; t < 16; t += 4) {
      const int tm = t >> 3, tn = t & 7;
      v8f acc = {};
#pragma unroll
      for (int kb = 0; kb < FEAT; kb += 32) {
        v16h a = load_a(&sm.Xs[0][0],  LF, tm * 16, kb, lane);
        v16h b = load_b(&sm.W1h[0][0], LF, tn * 16, kb, lane);
        acc = wmma16(a, b, acc);
      }
      const int g = lane >> 4, n = tn * 16 + (lane & 15);
#pragma unroll
      for (int v = 0; v < 8; ++v) {
        const int m = tm * 16 + g * 8 + v;
        const float z = acc[v] + sm.b1f[n];
        sm.u.tr.hS[m][n] = (_Float16)((z > 0.f) ? z : 0.f);
      }
    }
    __syncthreads();

    // (2) logits = hS @ W2^T + b2 : 2x1 tiles, K=128  (WMMA)
    for (int t = wave; t < 2; t += 4) {
      v8f acc = {};
#pragma unroll
      for (int kb = 0; kb < HID; kb += 32) {
        v16h a = load_a(&sm.u.tr.hS[0][0], LH, t * 16, kb, lane);
        v16h b = load_b(&sm.W2h[0][0],     LH, 0,      kb, lane);
        acc = wmma16(a, b, acc);
      }
      const int g = lane >> 4, n = lane & 15;
#pragma unroll
      for (int v = 0; v < 8; ++v)
        sm.u.tr.logits[t * 16 + g * 8 + v][n] = acc[v] + sm.b2f[n];
    }
    __syncthreads();

    // (3) softmax + dlogits (one thread per support row)
    if (tid < S_SUP) {
      float mx = -1e30f;
#pragma unroll
      for (int w = 0; w < WAY; ++w) mx = fmaxf(mx, sm.u.tr.logits[tid][w]);
      float ex[WAY], sum = 0.f;
#pragma unroll
      for (int w = 0; w < WAY; ++w) {
        ex[w] = __expf(sm.u.tr.logits[tid][w] - mx);
        sum += ex[w];
      }
      const float inv = 1.0f / (sum * (float)S_SUP);
      const int   tg  = sm.tgt[tid];
#pragma unroll
      for (int w = 0; w < WP; ++w)
        sm.u.tr.dlog[tid][w] =
            (w < WAY) ? (ex[w] * inv - ((w == tg) ? (1.0f / S_SUP) : 0.f)) : 0.f;
    } else if (tid < SP) {
#pragma unroll
      for (int w = 0; w < WP; ++w) sm.u.tr.dlog[tid][w] = 0.f;
    }
    __syncthreads();

    // (4) dz1 = (dlog @ W2) * relu'(z1), stored transposed (K=5: VALU)
    //     lanes walk s -> conflict-free b16 stores into dz1T columns
    for (int i = tid; i < SP * HID; i += 128) {
      const int s = i & 31, j = i >> 5;
      float dh = 0.f;
#pragma unroll
      for (int w = 0; w < WAY; ++w) dh += sm.u.tr.dlog[s][w] * sm.W2f[w][j];
      const float hv = (float)sm.u.tr.hS[s][j];
      sm.u.tr.dz1T[j][s] = (_Float16)((hv > 0.f) ? dh : 0.f);
    }
    __syncthreads();

    // (5) W2/b2 update (M=5: VALU)
    for (int i = tid; i < WAY * HID; i += 128) {
      const int w = i / HID, j = i % HID;
      float g2 = 0.f;
      for (int s = 0; s < S_SUP; ++s)
        g2 += sm.u.tr.dlog[s][w] * (float)sm.u.tr.hS[s][j];
      const float nv = sm.W2f[w][j] - LR * g2;
      sm.W2f[w][j] = nv;
      sm.W2h[w][j] = (_Float16)nv;
    }
    if (tid < WAY) {
      float gb = 0.f;
      for (int s = 0; s < S_SUP; ++s) gb += sm.u.tr.dlog[s][tid];
      sm.b2f[tid] -= LR * gb;
    }
    // (6) b1 update (one hidden unit per thread)
    {
      float gb = 0.f;
      for (int s = 0; s < S_SUP; ++s) gb += (float)sm.u.tr.dz1T[tid][s];
      sm.b1f[tid] -= LR * gb;
    }

    // (7) W1 -= LR * (dz1^T @ Xs) : 8x4 tiles, K=32 (padded)  (WMMA)
    for (int t = wave; t < 32; t += 4) {
      const int tm = t >> 2, tn = t & 3;
      v8f acc = {};
      v16h a = load_a(&sm.u.tr.dz1T[0][0], LS, tm * 16, 0, lane);
      v16h b = load_b(&sm.XsT[0][0],       LS, tn * 16, 0, lane);
      acc = wmma16(a, b, acc);
      const int g = lane >> 4, n = tn * 16 + (lane & 15);
#pragma unroll
      for (int v = 0; v < 8; ++v) {
        const int m = tm * 16 + g * 8 + v;
        const float nv = sm.W1f[m][n] - LR * acc[v];
        sm.W1f[m][n] = nv;
        sm.W1h[m][n] = (_Float16)nv;
      }
    }
    __syncthreads();
  }

  // ---------------- Query forward ----------------
  // hQ = relu(Xq @ W1^T + b1) : 5x8 tiles, K=64  (WMMA; overlaps dead scratch)
  for (int t = wave; t < 40; t += 4) {
    const int tm = t / 8, tn = t % 8;
    v8f acc = {};
#pragma unroll
    for (int kb = 0; kb < FEAT; kb += 32) {
      v16h a = load_a(&sm.Xq[0][0],  LF, tm * 16, kb, lane);
      v16h b = load_b(&sm.W1h[0][0], LF, tn * 16, kb, lane);
      acc = wmma16(a, b, acc);
    }
    const int g = lane >> 4, n = tn * 16 + (lane & 15);
#pragma unroll
    for (int v = 0; v < 8; ++v) {
      const int m = tm * 16 + g * 8 + v;
      const float z = acc[v] + sm.b1f[n];
      sm.u.hQ[m][n] = (_Float16)((z > 0.f) ? z : 0.f);
    }
  }
  __syncthreads();

  // out = hQ @ W2^T + b2 : 5x1 tiles, K=128  (WMMA), scatter valid lanes
  for (int t = wave; t < 5; t += 4) {
    v8f acc = {};
#pragma unroll
    for (int kb = 0; kb < HID; kb += 32) {
      v16h a = load_a(&sm.u.hQ[0][0], LH, t * 16, kb, lane);
      v16h b = load_b(&sm.W2h[0][0],  LH, 0,      kb, lane);
      acc = wmma16(a, b, acc);
    }
    const int g = lane >> 4, n = lane & 15;
    if (n < WAY) {
#pragma unroll
      for (int v = 0; v < 8; ++v) {
        const int m = t * 16 + g * 8 + v;
        if (m < E_Q)
          out[((size_t)e * E_Q + m) * WAY + n] = acc[v] + sm.b2f[n];
      }
    }
  }
}

extern "C" void kernel_launch(void* const* d_in, const int* in_sizes, int n_in,
                              void* d_out, int out_size, void* d_ws, size_t ws_size,
                              hipStream_t stream) {
  const float* qf  = (const float*)d_in[0];  // [E, Q, FEAT]
  const float* sf  = (const float*)d_in[1];  // [E, S, FEAT]
  const int*   st  = (const int*)d_in[2];    // [E, S]
  const float* W1  = (const float*)d_in[3];  // [E, HID, FEAT]
  const float* b1  = (const float*)d_in[4];  // [E, HID]
  const float* W2  = (const float*)d_in[5];  // [E, WAY, HID]
  const float* b2  = (const float*)d_in[6];  // [E, WAY]
  float* out = (float*)d_out;

  const int E = in_sizes[2] / S_SUP;         // 4000
  const size_t shmem = sizeof(SharedMem);    // ~101 KB -> 3 WGs / WGP

  episode_train_eval_kernel<<<dim3(E), dim3(128), shmem, stream>>>(
      qf, sf, st, W1, b1, W2, b2, out);
}
